// Explainer_HGNN_67370857005117
// MI455X (gfx1250) — matrix operand order repaired and checked by reading, not compiled, not run on hardware
//
#include <hip/hip_runtime.h>
#include <hip/hip_bf16.h>
#include <stdint.h>

typedef __attribute__((ext_vector_type(16))) _Float16 v16h;
typedef __attribute__((ext_vector_type(8)))  float    v8f;

#define DH 64
#define NGRAPH 64

__device__ __forceinline__ unsigned enc_f32(float x) {
    unsigned b = __float_as_uint(x);
    return (b & 0x80000000u) ? ~b : (b | 0x80000000u);
}
__device__ __forceinline__ float dec_f32(unsigned u) {
    return (u & 0x80000000u) ? __uint_as_float(u ^ 0x80000000u)
                             : __uint_as_float(~u);
}

// ---------------- init: zero counts, seed logits with bm, reset softmax state
__global__ void init_kernel(int* __restrict__ counts, float* __restrict__ logits,
                            unsigned* __restrict__ gmaxu, float* __restrict__ gsum,
                            const float* __restrict__ bm, int N, int E) {
    int i = blockIdx.x * blockDim.x + threadIdx.x;
    if (i < N) counts[i] = 0;
    if (i < E) logits[i] = bm[0];
    if (i < NGRAPH) { gmaxu[i] = 0u; gsum[i] = 0.f; }
}

// ---------------- node endpoint counts + per-edge graph id
__global__ void count_kernel(const int* __restrict__ ei, const int* __restrict__ batch,
                             int* __restrict__ counts, int* __restrict__ eb, int E) {
    int e = blockIdx.x * blockDim.x + threadIdx.x;
    if (e >= E) return;
    int s = ei[e], d = ei[E + e];
    atomicAdd(&counts[s], 1);
    atomicAdd(&counts[d], 1);
    eb[e] = batch[s];
}

// ---------------- ea = 0.5*(x[src] + x[dst]),  16 threads/edge, float4 each
__global__ void ea_kernel(const float* __restrict__ x, const int* __restrict__ ei,
                          float* __restrict__ H, int E) {
    size_t t = (size_t)blockIdx.x * blockDim.x + threadIdx.x;
    int e = (int)(t >> 4);
    if (e >= E) return;
    int d4 = (int)(t & 15) << 2;
    int s = ei[e], d = ei[E + e];
    float4 a = *(const float4*)(x + (size_t)s * DH + d4);
    float4 b = *(const float4*)(x + (size_t)d * DH + d4);
    float4 r;
    r.x = 0.5f * (a.x + b.x); r.y = 0.5f * (a.y + b.y);
    r.z = 0.5f * (a.z + b.z); r.w = 0.5f * (a.w + b.w);
    *(float4*)(H + (size_t)e * DH + d4) = r;
}

// ---------------- W[k][n] (f32) -> Wt[l][n][k] (f16, transposed for B-frag reads)
__global__ void convw_kernel(const float* __restrict__ W0, const float* __restrict__ W1,
                             const float* __restrict__ W2, _Float16* __restrict__ Wt) {
    int t = blockIdx.x * blockDim.x + threadIdx.x;
    if (t >= 3 * 4096) return;
    int l = t >> 12, rem = t & 4095, k = rem >> 6, n = rem & 63;
    const float* W = (l == 0) ? W0 : ((l == 1) ? W1 : W2);
    Wt[l * 4096 + n * 64 + k] = (_Float16)W[k * 64 + n];
}

// ---------------- XL = H @ W   via v_wmma_f32_16x16x32_f16
// block = 128 threads = 4 waves; each wave computes a 32x64 output slab
// (2 M-tiles x 4 N-tiles, K=64 in two K=32 steps -> 16 WMMAs/wave).
__global__ void __launch_bounds__(128) gemm_kernel(const float* __restrict__ Hin,
                                                   const _Float16* __restrict__ Wt,
                                                   float* __restrict__ XL, int E) {
    __shared__ _Float16 lw[64 * 64];   // W^T (n-major) in f16, 8 KB
    {
        const uint32_t* wg = (const uint32_t*)Wt;
        uint32_t* wl = (uint32_t*)lw;
        for (int i = threadIdx.x; i < 2048; i += 128) wl[i] = wg[i];
    }
    __syncthreads();

    int wave = threadIdx.x >> 5;
    int lane = threadIdx.x & 31;
    int r0   = blockIdx.x * 128 + wave * 32;       // 32 rows per wave
    int kh8  = (lane >> 4) * 8;                    // A-frag K half-select
    int kh16 = (lane >> 4) * 16;                   // unused; B-frag reads use kh8
    bool full = (r0 + 32 <= E);                    // wave-uniform tail check

    // A fragments: [mt][kt]; 16x32 f16 layout
    v16h a[2][2];
#pragma unroll
    for (int mt = 0; mt < 2; ++mt) {
        int mrow = r0 + mt * 16 + (lane & 15);
        if (!full && mrow >= E) mrow = E - 1;      // clamp keeps EXEC all-ones
        const float* hrow = Hin + (size_t)mrow * DH;
#pragma unroll
        for (int kt = 0; kt < 2; ++kt) {
            int base = 32 * kt + kh8;
            float4 q0 = *(const float4*)(hrow + base);
            float4 q1 = *(const float4*)(hrow + base + 4);
            float4 q2 = *(const float4*)(hrow + base + 16);
            float4 q3 = *(const float4*)(hrow + base + 20);
            v16h v;
            v[0]  = (_Float16)q0.x; v[1]  = (_Float16)q0.y;
            v[2]  = (_Float16)q0.z; v[3]  = (_Float16)q0.w;
            v[4]  = (_Float16)q1.x; v[5]  = (_Float16)q1.y;
            v[6]  = (_Float16)q1.z; v[7]  = (_Float16)q1.w;
            v[8]  = (_Float16)q2.x; v[9]  = (_Float16)q2.y;
            v[10] = (_Float16)q2.z; v[11] = (_Float16)q2.w;
            v[12] = (_Float16)q3.x; v[13] = (_Float16)q3.y;
            v[14] = (_Float16)q3.z; v[15] = (_Float16)q3.w;
            a[mt][kt] = v;
        }
    }

#pragma unroll
    for (int nt = 0; nt < 4; ++nt) {
        int col = nt * 16 + (lane & 15);           // B-frag N / store column
        // B fragments for this N-tile (reused across both M-tiles)
        v16h b[2];
#pragma unroll
        for (int kt = 0; kt < 2; ++kt) {
            const _Float16* wr = lw + col * 64 + kt * 32 + kh8;   // 2 runs of 8 f16 (k, k+16)
            v16h bv;
#pragma unroll
            for (int i = 0; i < 16; ++i) bv[i] = wr[(i & 7) + (i >> 3) * 16];
            b[kt] = bv;
        }
#pragma unroll
        for (int mt = 0; mt < 2; ++mt) {
            v8f c = {};
            c = __builtin_amdgcn_wmma_f32_16x16x32_f16(
                    false, a[mt][0], false, b[0], (short)0, c, false, false);
            c = __builtin_amdgcn_wmma_f32_16x16x32_f16(
                    false, a[mt][1], false, b[1], (short)0, c, false, false);
            int mb = r0 + mt * 16 + (lane >> 4) * 8;   // C layout: VGPR r -> M = r (+8 hi)
            float* outp = XL + (size_t)mb * DH + col;
            if (full) {
#pragma unroll
                for (int r = 0; r < 8; ++r) outp[(size_t)r * DH] = c[r];
            } else {
#pragma unroll
                for (int r = 0; r < 8; ++r)
                    if (mb + r < E) outp[(size_t)r * DH] = c[r];
            }
        }
    }
}

__global__ void zero_kernel(float* __restrict__ p, int n) {
    int i = blockIdx.x * blockDim.x + threadIdx.x;
    if (i < n) p[i] = 0.f;
}

// ---------------- m[node] += xl[e] for both endpoints (L2-resident f32 atomics)
__global__ void scatter_kernel(const float* __restrict__ XL, const int* __restrict__ ei,
                               float* __restrict__ M, int E) {
    size_t t = (size_t)blockIdx.x * blockDim.x + threadIdx.x;
    int e = (int)(t >> 4);
    if (e >= E) return;
    int d4 = (int)(t & 15) << 2;
    float4 v = *(const float4*)(XL + (size_t)e * DH + d4);
    int s = ei[e], d = ei[E + e];
    float* ms = M + (size_t)s * DH + d4;
    atomicAdd(ms + 0, v.x); atomicAdd(ms + 1, v.y);
    atomicAdd(ms + 2, v.z); atomicAdd(ms + 3, v.w);
    float* md = M + (size_t)d * DH + d4;
    atomicAdd(md + 0, v.x); atomicAdd(md + 1, v.y);
    atomicAdd(md + 2, v.z); atomicAdd(md + 3, v.w);
}

// ---------------- o = relu((cs*m[s] + cd*m[d] + xl)/Dg + b); logits += o . Wm_l
__global__ void gather_kernel(const float* __restrict__ XL, const float* __restrict__ M,
                              const int* __restrict__ ei, const int* __restrict__ counts,
                              const float* __restrict__ bias, const float* __restrict__ WmSeg,
                              float* __restrict__ Hout, float* __restrict__ logits, int E) {
    size_t t = (size_t)blockIdx.x * blockDim.x + threadIdx.x;
    int e = (int)(t >> 4);
    if (e >= E) return;
    int d4 = (int)(t & 15) << 2;
    int s = ei[e], d = ei[E + e];
    int cs = counts[s], cd = counts[d];
    float ws = (cs != 1) ? 1.f : 0.f;
    float wd = (cd != 1) ? 1.f : 0.f;
    float dinv  = 1.f / (ws + wd + 1.f);
    float coefs = (cs >= 2) ? (1.f / (float)cs) : 0.f;
    float coefd = (cd >= 2) ? (1.f / (float)cd) : 0.f;

    float4 xv  = *(const float4*)(XL + (size_t)e * DH + d4);
    float4 msv = *(const float4*)(M + (size_t)s * DH + d4);
    float4 mdv = *(const float4*)(M + (size_t)d * DH + d4);
    float4 bv  = *(const float4*)(bias + d4);
    float4 o;
    o.x = fmaxf((coefs * msv.x + coefd * mdv.x + xv.x) * dinv + bv.x, 0.f);
    o.y = fmaxf((coefs * msv.y + coefd * mdv.y + xv.y) * dinv + bv.y, 0.f);
    o.z = fmaxf((coefs * msv.z + coefd * mdv.z + xv.z) * dinv + bv.z, 0.f);
    o.w = fmaxf((coefs * msv.w + coefd * mdv.w + xv.w) * dinv + bv.w, 0.f);
    *(float4*)(Hout + (size_t)e * DH + d4) = o;

    float4 wv = *(const float4*)(WmSeg + d4);
    float part = o.x * wv.x + o.y * wv.y + o.z * wv.z + o.w * wv.w;
#pragma unroll
    for (int off = 8; off >= 1; off >>= 1) part += __shfl_down(part, off, 16);
    if ((t & 15) == 0) atomicAdd(&logits[e], part);
}

// ---------------- segment softmax over 64 graphs
__global__ void smax_kernel(const float* __restrict__ logits, const int* __restrict__ eb,
                            unsigned* __restrict__ gmaxu, int E) {
    int i = blockIdx.x * blockDim.x + threadIdx.x;
    if (i < E) atomicMax(&gmaxu[eb[i]], enc_f32(logits[i]));
}
__global__ void expsum_kernel(const float* __restrict__ logits, const int* __restrict__ eb,
                              const unsigned* __restrict__ gmaxu, float* __restrict__ gsum,
                              float* __restrict__ out, int E) {
    int i = blockIdx.x * blockDim.x + threadIdx.x;
    if (i >= E) return;
    float mx = dec_f32(gmaxu[eb[i]]);
    float z = __expf(logits[i] - mx);
    out[i] = z;
    atomicAdd(&gsum[eb[i]], z);
}
__global__ void div_kernel(float* __restrict__ out, const int* __restrict__ eb,
                           const float* __restrict__ gsum, int E) {
    int i = blockIdx.x * blockDim.x + threadIdx.x;
    if (i < E) out[i] = out[i] / gsum[eb[i]];
}

extern "C" void kernel_launch(void* const* d_in, const int* in_sizes, int n_in,
                              void* d_out, int out_size, void* d_ws, size_t ws_size,
                              hipStream_t stream) {
    const float* x     = (const float*)d_in[0];
    const int*   ei    = (const int*)d_in[1];
    const int*   batch = (const int*)d_in[3];
    const float* W[3]  = {(const float*)d_in[4], (const float*)d_in[6], (const float*)d_in[8]};
    const float* B[3]  = {(const float*)d_in[5], (const float*)d_in[7], (const float*)d_in[9]};
    const float* Wm    = (const float*)d_in[10];
    const float* bm    = (const float*)d_in[11];
    int N = in_sizes[0] / DH;
    int E = in_sizes[1] / 2;
    float* out = (float*)d_out;

    char* p = (char*)d_ws;
    auto alloc = [&](size_t bytes) -> char* {
        char* r = p;
        p += (bytes + 255) & ~(size_t)255;
        return r;
    };
    float*     Hbuf   = (float*)alloc((size_t)E * DH * 4);
    float*     XL     = (float*)alloc((size_t)E * DH * 4);
    float*     M      = (float*)alloc((size_t)N * DH * 4);
    int*       counts = (int*)alloc((size_t)N * 4);
    int*       eb     = (int*)alloc((size_t)E * 4);
    float*     logits = (float*)alloc((size_t)E * 4);
    _Float16*  Wt     = (_Float16*)alloc(3 * 4096 * 2);
    unsigned*  gmaxu  = (unsigned*)alloc(NGRAPH * 4);
    float*     gsum   = (float*)alloc(NGRAPH * 4);

    const int TB = 256;
    int    gE  = (E + TB - 1) / TB;
    size_t ew  = (size_t)E * 16;
    int    gEw = (int)((ew + TB - 1) / TB);
    int    gM  = (N * DH + TB - 1) / TB;

    init_kernel<<<gE, TB, 0, stream>>>(counts, logits, gmaxu, gsum, bm, N, E);
    count_kernel<<<gE, TB, 0, stream>>>(ei, batch, counts, eb, E);
    ea_kernel<<<gEw, TB, 0, stream>>>(x, ei, Hbuf, E);
    convw_kernel<<<(3 * 4096 + TB - 1) / TB, TB, 0, stream>>>(W[0], W[1], W[2], Wt);

    for (int l = 0; l < 3; ++l) {
        gemm_kernel<<<(E + 127) / 128, 128, 0, stream>>>(Hbuf, Wt + l * 4096, XL, E);
        zero_kernel<<<gM, TB, 0, stream>>>(M, N * DH);
        scatter_kernel<<<gEw, TB, 0, stream>>>(XL, ei, M, E);
        gather_kernel<<<gEw, TB, 0, stream>>>(XL, M, ei, counts, B[l], Wm + l * DH,
                                              Hbuf, logits, E);
    }

    smax_kernel<<<gE, TB, 0, stream>>>(logits, eb, gmaxu, E);
    expsum_kernel<<<gE, TB, 0, stream>>>(logits, eb, gmaxu, gsum, out, E);
    div_kernel<<<gE, TB, 0, stream>>>(out, eb, gsum, E);
}
